// MambaBlock_57827439674254
// MI455X (gfx1250) — matrix operand order repaired and checked by reading, not compiled, not run on hardware
//
#include <hip/hip_runtime.h>
#include <math.h>

typedef __attribute__((ext_vector_type(16))) _Float16 v16h;
typedef __attribute__((ext_vector_type(8)))  _Float16 v8h;
typedef __attribute__((ext_vector_type(4)))  _Float16 v4h;
typedef __attribute__((ext_vector_type(8)))  float    v8f;
typedef __attribute__((ext_vector_type(4)))  int      v4i;
typedef __attribute__((ext_vector_type(2)))  int      v2i;

#define DM   1024
#define DIN  2048
#define DST  16
#define RK   64
#define HID  2048
#define NB   4
#define LSEQ 2048
#define MROWS (NB*LSEQ)   // 8192 tokens

// ---- CDNA5 async global->LDS path (guarded; falls back to ds_store staging) ----
#if defined(__has_builtin)
#if __has_builtin(__builtin_amdgcn_global_load_async_to_lds_b128) && \
    __has_builtin(__builtin_amdgcn_global_load_async_to_lds_b64) &&  \
    __has_builtin(__builtin_amdgcn_s_wait_asynccnt)
#define USE_ASYNC 1
#endif
#endif

#ifdef USE_ASYNC
typedef __attribute__((address_space(1))) v4i* gp4;
typedef __attribute__((address_space(3))) v4i* lp4;
typedef __attribute__((address_space(1))) v2i* gp2;
typedef __attribute__((address_space(3))) v2i* lp2;
__device__ __forceinline__ void async_cp128(const void* g, void* l) {
  __builtin_amdgcn_global_load_async_to_lds_b128((gp4)(void*)g, (lp4)l, 0, 0);
}
__device__ __forceinline__ void async_cp64(const void* g, void* l) {
  __builtin_amdgcn_global_load_async_to_lds_b64((gp2)(void*)g, (lp2)l, 0, 0);
}
#endif

// ---------------- weight convert + transpose: Wt[k*N+n] = (f16)W[n*K+k] ----------------
__global__ __launch_bounds__(256) void k_cvt_tr(const float* __restrict__ W,
                                                _Float16* __restrict__ Wt, int N, int K) {
  int i = blockIdx.x * 256 + threadIdx.x;
  if (i < N * K) {
    int n = i / K, k = i - n * K;
    Wt[(size_t)k * N + n] = (_Float16)W[i];
  }
}

// ---------------- LayerNorm (row=1024) f32 -> f16 ----------------
__global__ __launch_bounds__(256) void k_ln(const float* __restrict__ x,
                                            const float* __restrict__ g,
                                            const float* __restrict__ b,
                                            _Float16* __restrict__ out) {
  __shared__ float s1[256], s2[256];
  int row = blockIdx.x, t = threadIdx.x;
  const float* xr = x + (size_t)row * DM;
  float v[4], sum = 0.f, sq = 0.f;
#pragma unroll
  for (int i = 0; i < 4; ++i) { v[i] = xr[t * 4 + i]; sum += v[i]; sq += v[i] * v[i]; }
  s1[t] = sum; s2[t] = sq; __syncthreads();
  for (int off = 128; off > 0; off >>= 1) {
    if (t < off) { s1[t] += s1[t + off]; s2[t] += s2[t + off]; }
    __syncthreads();
  }
  float m   = s1[0] * (1.f / DM);
  float var = s2[0] * (1.f / DM) - m * m;
  float r   = rsqrtf(var + 1e-5f);
  _Float16* orow = out + (size_t)row * DM;
#pragma unroll
  for (int i = 0; i < 4; ++i) {
    int c = t * 4 + i;
    orow[c] = (_Float16)((v[i] - m) * r * g[c] + b[c]);
  }
}

// ---------------- causal depthwise conv(4) + SiLU ----------------
__global__ __launch_bounds__(256) void k_conv_silu(const _Float16* __restrict__ xz,
                                                   const float* __restrict__ cw,
                                                   const float* __restrict__ cb,
                                                   _Float16* __restrict__ xa) {
  int i = blockIdx.x * 256 + threadIdx.x;     // over NB*LSEQ*DIN
  int d  = i & (DIN - 1);
  int l  = (i >> 11) & (LSEQ - 1);
  int bq = i >> 22;
  float acc = cb[d];
#pragma unroll
  for (int k = 0; k < 4; ++k) {
    int ll = l - 3 + k;
    if (ll >= 0)
      acc += (float)xz[(size_t)(bq * LSEQ + ll) * (2 * DIN) + d] * cw[d * 4 + k];
  }
  float s = acc / (1.f + __expf(-acc));
  xa[(size_t)i] = (_Float16)s;
}

// ---------------- selective scan: one lane per (b,d); B/C broadcast via shuffles ----------------
__global__ __launch_bounds__(256) void k_scan(const float* __restrict__ delta,
                                              const _Float16* __restrict__ xa,
                                              const _Float16* __restrict__ xdbl,
                                              const _Float16* __restrict__ xz,
                                              const float* __restrict__ A_log,
                                              const float* __restrict__ Dvec,
                                              _Float16* __restrict__ y16) {
  int d = blockIdx.x * 256 + threadIdx.x;
  int b = blockIdx.y;
  int lane = threadIdx.x & 31;
  float Ar[DST];
#pragma unroll
  for (int n = 0; n < DST; ++n) Ar[n] = -__expf(A_log[d * DST + n]);
  float Dd = Dvec[d];
  float h[DST];
#pragma unroll
  for (int n = 0; n < DST; ++n) h[n] = 0.f;

  for (int l = 0; l < LSEQ; ++l) {
    size_t tok = (size_t)(b * LSEQ + l);
    // lane k holds B[k] (k<16) or C[k-16] (k>=16); broadcast with wave shuffles
    float bcv = (float)xdbl[tok * 96 + 64 + lane];
    float dt  = delta[tok * DIN + d];
    float u   = (float)xa[tok * DIN + d];
    float du  = dt * u;
    float y   = 0.f;
#pragma unroll
    for (int n = 0; n < DST; ++n) {
      float dA = __expf(dt * Ar[n]);
      h[n] = h[n] * dA + du * __shfl(bcv, n);
      y   += h[n] * __shfl(bcv, 16 + n);
    }
    y += u * Dd;
    float z = (float)xz[tok * (2 * DIN) + DIN + d];
    y *= z / (1.f + __expf(-z));
    y16[tok * DIN + d] = (_Float16)y;
  }
}

// ---------------- WMMA GEMM: out[M,N] = A[M,K] * Wt[K,N] ----------------
// Block: 256 thr = 8 waves stacked along M. Block tile 256(M) x 32(N).
// Wave tile 32x32 = 2x2 WMMA frags. A/B panels staged in LDS (async double-buffered).
enum { EPI_F16 = 0, EPI_SOFTPLUS = 1, EPI_RESID = 2, EPI_BIAS_GELU = 3, EPI_BIAS_RESID = 4 };

template <int EPI>
__global__ __launch_bounds__(256) void k_gemm(const _Float16* __restrict__ A, int lda,
                                              const _Float16* __restrict__ Wt, int ldw,
                                              float* __restrict__ outF,
                                              _Float16* __restrict__ outH, int ldc,
                                              const float* __restrict__ bias,
                                              const float* __restrict__ resid, int K) {
  __shared__ _Float16 sA[2][256 * 32];   // 32 KB
  __shared__ _Float16 sB[2][32 * 32];    //  4 KB
  int tid  = threadIdx.x;
  int lane = tid & 31, wave = tid >> 5;
  int mblk = blockIdx.y * 256;
  int nblk = blockIdx.x * 32;
  int mr = lane & 15, hi = lane >> 4;
  int kk = mr + hi * 16;                 // B-frag K row for this lane

  v8f acc[2][2] = {};

  auto stage = [&](int s) {
    int k0 = s * 32, buf = s & 1;
#pragma unroll
    for (int q = 0; q < 4; ++q) {        // A: 256x32 halves = 1024 16B-chunks
      int c = q * 256 + tid, r = c >> 2, c16 = c & 3;
      const _Float16* g = A + (size_t)(mblk + r) * lda + k0 + c16 * 8;
      _Float16* lp = &sA[buf][r * 32 + c16 * 8];
#ifdef USE_ASYNC
      async_cp128(g, lp);
#else
      *(v8h*)lp = *(const v8h*)g;
#endif
    }
    {                                    // B: 32x32 halves = 256 8B-chunks
      int r = tid >> 3, c8 = tid & 7;
      const _Float16* g = Wt + (size_t)(k0 + r) * ldw + nblk + c8 * 4;
      _Float16* lp = &sB[buf][r * 32 + c8 * 4];
#ifdef USE_ASYNC
      async_cp64(g, lp);
#else
      *(v4h*)lp = *(const v4h*)g;
#endif
    }
  };

  auto compute = [&](int buf) {
    union AU { v16h v; v8h h[2]; } a0, a1;
    int rb = wave * 32 + mr;
    a0.h[0] = *(const v8h*)&sA[buf][(rb)*32 + hi * 8];
    a0.h[1] = *(const v8h*)&sA[buf][(rb)*32 + 16 + hi * 8];
    a1.h[0] = *(const v8h*)&sA[buf][(rb + 16) * 32 + hi * 8];
    a1.h[1] = *(const v8h*)&sA[buf][(rb + 16) * 32 + 16 + hi * 8];
    v16h b0, b1;
#pragma unroll
    for (int s = 0; s < 16; ++s) {
      int k = (s < 8) ? hi * 8 + s : 16 + hi * 8 + (s - 8);
      b0[s] = sB[buf][k * 32 + mr];
      b1[s] = sB[buf][k * 32 + 16 + mr];
    }
    acc[0][0] = __builtin_amdgcn_wmma_f32_16x16x32_f16(false, a0.v, false, b0, (short)0, acc[0][0], false, false);
    acc[0][1] = __builtin_amdgcn_wmma_f32_16x16x32_f16(false, a0.v, false, b1, (short)0, acc[0][1], false, false);
    acc[1][0] = __builtin_amdgcn_wmma_f32_16x16x32_f16(false, a1.v, false, b0, (short)0, acc[1][0], false, false);
    acc[1][1] = __builtin_amdgcn_wmma_f32_16x16x32_f16(false, a1.v, false, b1, (short)0, acc[1][1], false, false);
  };

  int S = K / 32;
#ifdef USE_ASYNC
  stage(0);
  for (int s = 0; s < S; ++s) {
    if (s + 1 < S) {
      stage(s + 1);
      __builtin_amdgcn_s_wait_asynccnt(5);  // 5 async ops/wave just issued; buf s done
    } else {
      __builtin_amdgcn_s_wait_asynccnt(0);
    }
    __syncthreads();
    compute(s & 1);
    __syncthreads();
  }
#else
  for (int s = 0; s < S; ++s) {
    stage(s);
    __syncthreads();
    compute(s & 1);
    __syncthreads();
  }
#endif

#pragma unroll
  for (int i = 0; i < 2; ++i) {
#pragma unroll
    for (int j = 0; j < 2; ++j) {
      union CU { v8f v; float f[8]; } cu; cu.v = acc[i][j];
      int col = nblk + j * 16 + mr;
      float bval = (EPI == EPI_SOFTPLUS || EPI == EPI_BIAS_GELU || EPI == EPI_BIAS_RESID)
                       ? bias[col] : 0.f;
#pragma unroll
      for (int r = 0; r < 8; ++r) {
        int m = mblk + wave * 32 + i * 16 + r + hi * 8;  // C/D: VGPR r -> M=r(+8 for hi lanes)
        size_t o = (size_t)m * ldc + col;
        float v = cu.f[r];
        if constexpr (EPI == EPI_F16) {
          outH[o] = (_Float16)v;
        } else if constexpr (EPI == EPI_SOFTPLUS) {
          float t = v + bval;
          outF[o] = (t > 20.f) ? t : log1pf(__expf(t));
        } else if constexpr (EPI == EPI_RESID) {
          outF[o] = v + resid[o];
        } else if constexpr (EPI == EPI_BIAS_GELU) {
          float t = v + bval;
          outH[o] = (_Float16)(0.5f * t * (1.f + erff(t * 0.70710678118f)));
        } else {  // EPI_BIAS_RESID
          outF[o] = v + bval + resid[o];
        }
      }
    }
  }
}

extern "C" void kernel_launch(void* const* d_in, const int* in_sizes, int n_in,
                              void* d_out, int out_size, void* d_ws, size_t ws_size,
                              hipStream_t stream) {
  const float* x          = (const float*)d_in[0];
  const float* ln1_g      = (const float*)d_in[1];
  const float* ln1_b      = (const float*)d_in[2];
  const float* in_proj_w  = (const float*)d_in[3];
  const float* conv_w     = (const float*)d_in[4];
  const float* conv_b     = (const float*)d_in[5];
  const float* x_proj_w   = (const float*)d_in[6];
  const float* dt_proj_w  = (const float*)d_in[7];
  const float* dt_proj_b  = (const float*)d_in[8];
  const float* A_log      = (const float*)d_in[9];
  const float* Dv         = (const float*)d_in[10];
  const float* out_proj_w = (const float*)d_in[11];
  const float* ln2_g      = (const float*)d_in[12];
  const float* ln2_b      = (const float*)d_in[13];
  const float* mlp_w1     = (const float*)d_in[14];
  const float* mlp_b1     = (const float*)d_in[15];
  const float* mlp_w2     = (const float*)d_in[16];
  const float* mlp_b2     = (const float*)d_in[17];
  float* out = (float*)d_out;

  char* ws = (char*)d_ws;
  size_t off = 0;
  auto wsalloc = [&](size_t bytes) -> char* {
    char* p = ws + off;
    off = (off + bytes + 255) & ~(size_t)255;
    return p;
  };
  _Float16* winT   = (_Float16*)wsalloc((size_t)DM * (2 * DIN) * 2);
  _Float16* wxpT   = (_Float16*)wsalloc((size_t)DIN * 96 * 2);
  _Float16* wdtT   = (_Float16*)wsalloc((size_t)RK * DIN * 2);
  _Float16* woutT  = (_Float16*)wsalloc((size_t)DIN * DM * 2);
  _Float16* w1T    = (_Float16*)wsalloc((size_t)DM * HID * 2);
  _Float16* w2T    = (_Float16*)wsalloc((size_t)HID * DM * 2);
  _Float16* h16    = (_Float16*)wsalloc((size_t)MROWS * DM * 2);
  _Float16* xz16   = (_Float16*)wsalloc((size_t)MROWS * 2 * DIN * 2);
  _Float16* xa16   = (_Float16*)wsalloc((size_t)MROWS * DIN * 2);
  _Float16* xdbl16 = (_Float16*)wsalloc((size_t)MROWS * 96 * 2);
  float*    delta  = (float*)   wsalloc((size_t)MROWS * DIN * 4);
  _Float16* y16    = (_Float16*)wsalloc((size_t)MROWS * DIN * 2);
  float*    x2     = (float*)   wsalloc((size_t)MROWS * DM * 4);
  _Float16* m116   = (_Float16*)wsalloc((size_t)MROWS * HID * 2);
  (void)ws_size; (void)in_sizes; (void)n_in; (void)out_size;

  dim3 blk(256);
  auto cdiv = [](int a, int b) { return (a + b - 1) / b; };

  // weights -> transposed f16
  k_cvt_tr<<<cdiv(2 * DIN * DM, 256), blk, 0, stream>>>(in_proj_w, winT, 2 * DIN, DM);
  k_cvt_tr<<<cdiv(96 * DIN, 256),     blk, 0, stream>>>(x_proj_w,  wxpT, 96, DIN);
  k_cvt_tr<<<cdiv(DIN * RK, 256),     blk, 0, stream>>>(dt_proj_w, wdtT, DIN, RK);
  k_cvt_tr<<<cdiv(DM * DIN, 256),     blk, 0, stream>>>(out_proj_w, woutT, DM, DIN);
  k_cvt_tr<<<cdiv(HID * DM, 256),     blk, 0, stream>>>(mlp_w1, w1T, HID, DM);
  k_cvt_tr<<<cdiv(DM * HID, 256),     blk, 0, stream>>>(mlp_w2, w2T, DM, HID);

  // LN1
  k_ln<<<MROWS, blk, 0, stream>>>(x, ln1_g, ln1_b, h16);

  // in_proj: (8192x1024) x (1024x4096) -> xz16
  k_gemm<EPI_F16><<<dim3((2 * DIN) / 32, MROWS / 256), blk, 0, stream>>>(
      h16, DM, winT, 2 * DIN, nullptr, xz16, 2 * DIN, nullptr, nullptr, DM);

  // conv + silu
  k_conv_silu<<<cdiv(NB * LSEQ * DIN, 256), blk, 0, stream>>>(xz16, conv_w, conv_b, xa16);

  // x_proj: (8192x2048) x (2048x96) -> xdbl16
  k_gemm<EPI_F16><<<dim3(96 / 32, MROWS / 256), blk, 0, stream>>>(
      xa16, DIN, wxpT, 96, nullptr, xdbl16, 96, nullptr, nullptr, DIN);

  // dt_proj + bias + softplus: (8192x64) x (64x2048) -> delta (f32)
  k_gemm<EPI_SOFTPLUS><<<dim3(DIN / 32, MROWS / 256), blk, 0, stream>>>(
      xdbl16, 96, wdtT, DIN, delta, nullptr, DIN, dt_proj_b, nullptr, RK);

  // selective scan (fused * silu(z)) -> y16
  k_scan<<<dim3(DIN / 256, NB), blk, 0, stream>>>(delta, xa16, xdbl16, xz16, A_log, Dv, y16);

  // out_proj + residual(x): -> x2
  k_gemm<EPI_RESID><<<dim3(DM / 32, MROWS / 256), blk, 0, stream>>>(
      y16, DIN, woutT, DM, x2, nullptr, DM, nullptr, x, DIN);

  // LN2 (reuse h16)
  k_ln<<<MROWS, blk, 0, stream>>>(x2, ln2_g, ln2_b, h16);

  // mlp1 + bias + exact GELU -> m116
  k_gemm<EPI_BIAS_GELU><<<dim3(HID / 32, MROWS / 256), blk, 0, stream>>>(
      h16, DM, w1T, HID, nullptr, m116, HID, mlp_b1, nullptr, DM);

  // mlp2 + bias + residual(x2) -> d_out
  k_gemm<EPI_BIAS_RESID><<<dim3(DM / 32, MROWS / 256), blk, 0, stream>>>(
      m116, HID, w2T, DM, out, nullptr, DM, mlp_b2, x2, HID);
}
